// TMTM_4698694222280
// MI455X (gfx1250) — matrix operand order repaired and checked by reading, not compiled, not run on hardware
//
#include <hip/hip_runtime.h>

// ---------------------------------------------------------------------------
// CDNA5 / gfx1250 fused RGCN. Dense math: v_wmma_f32_16x16x32_f16, one wave
// owns a 16x128 output tile (8 accumulators), weights pre-packed into
// B-fragment lane order. Activations X stored in f16 (halves the dominant
// gather/activation traffic; fp32 accumulate + fp32 LN stats keep numerics).
// Per-relation aggregation buffers stay fp32 (global_atomic_add_f32).
// Fusions: [branches -> LDS relayout -> in-proj], [root+rel0+rel1+bias+
// residual+LayerNorm], [out1+leaky+head].
// ---------------------------------------------------------------------------

typedef __attribute__((ext_vector_type(16))) _Float16 v16h;
typedef __attribute__((ext_vector_type(8)))  _Float16 v8h;
typedef __attribute__((ext_vector_type(4)))  _Float16 v4h;
typedef __attribute__((ext_vector_type(8)))  float    v8f;

#define HID 128
#define LDSW 132   // padded LDS row stride (floats): conflict-free A-frag reads
#define LEAKY(x) ((x) > 0.f ? (x) : 0.01f * (x))

// A fragment (16-bit A 16x32, ISA 7.12.2): lane L -> row = L&15,
// k0 = (L>>4)*8; halfs 0..7 = K[kb..kb+7], halfs 8..15 = K[kb+16..kb+23].

// fp32 source (feature matrix, LDS tile, fp32 agg): load + convert.
template <typename P>
__device__ __forceinline__ v16h load_a_frag_f32(P arow, int kb) {
    v16h a;
#pragma unroll
    for (int v = 0; v < 4; ++v) {
        a[2 * v]         = (_Float16)arow[kb + 2 * v];
        a[2 * v + 1]     = (_Float16)arow[kb + 2 * v + 1];
        a[8 + 2 * v]     = (_Float16)arow[kb + 16 + 2 * v];
        a[8 + 2 * v + 1] = (_Float16)arow[kb + 16 + 2 * v + 1];
    }
    return a;
}

// f16 source (activation matrix X): two 16-byte loads, no conversion.
__device__ __forceinline__ v16h load_a_frag_f16(const _Float16* __restrict__ arow, int kb) {
    v8h lo = *(const v8h*)(arow + kb);
    v8h hi = *(const v8h*)(arow + kb + 16);
    v16h a;
#pragma unroll
    for (int i = 0; i < 8; ++i) { a[i] = lo[i]; a[8 + i] = hi[i]; }
    return a;
}

__device__ __forceinline__ v16h load_b_frag(const _Float16* __restrict__ p) {
    return *(const v16h*)p;
}

#define WMMA(ACC, A, B) \
    (ACC) = __builtin_amdgcn_wmma_f32_16x16x32_f16(false, (A), false, (B), (short)0, (ACC), false, false)

// ---------------------------------------------------------------------------
// Pack fp32 W[K][Nc] into f16 WMMA B fragments (zero-padded K/N).
// Tile (kt,nt): lane L holds 16 contiguous halfs
//   b[j] = W[kt*32 + (L>>4)*8 + (j&7) + (j>>3)*16][nt*16 + (L&15)]
// ---------------------------------------------------------------------------
__global__ __launch_bounds__(256)
void pack_w_f16(const float* __restrict__ W, int K, int Nc, int Kt, int Nt,
                _Float16* __restrict__ dst) {
    int tid = blockIdx.x * blockDim.x + threadIdx.x;
    int total = Kt * Nt * 512;
    if (tid >= total) return;
    int j    = tid & 15;
    int lane = (tid >> 4) & 31;
    int blk  = tid >> 9;
    int nt = blk % Nt;
    int kt = blk / Nt;
    int col = nt * 16 + (lane & 15);
    int k   = kt * 32 + ((lane >> 4) << 3) + (j & 7) + ((j >> 3) << 4);
    float v = (k < K && col < Nc) ? W[(size_t)k * Nc + col] : 0.f;
    dst[tid] = (_Float16)v;
}

// ---------------------------------------------------------------------------
// Fused: 4 branch projections (leaky) -> 16x128 tile staged in LDS ->
// in-projection (leaky) -> X (f16). One wave per 16-row tile.
// ---------------------------------------------------------------------------
__global__ __launch_bounds__(128)
void fused_input(const float* __restrict__ F, int ldf,
                 const _Float16* __restrict__ des_p, const _Float16* __restrict__ tw_p,
                 const _Float16* __restrict__ num_p, const _Float16* __restrict__ cat_p,
                 const _Float16* __restrict__ in_p,
                 const float* __restrict__ des_b, const float* __restrict__ tw_b,
                 const float* __restrict__ num_b, const float* __restrict__ cat_b,
                 const float* __restrict__ in_b,
                 _Float16* __restrict__ X, int Mtiles) {
    __shared__ float xs[4 * 16 * LDSW];
    int wv   = threadIdx.x >> 5;
    int wave = blockIdx.x * 4 + wv;
    int lane = threadIdx.x & 31;
    if (wave >= Mtiles) return;          // wave-uniform
    int lc = lane & 15, hi = lane >> 4, k0 = hi * 8;
    float* ls = xs + wv * 16 * LDSW;

    const float* arow = F + (size_t)(wave * 16 + lc) * ldf;
    v8f acc[8] = {};

    const _Float16* wps[4] = {des_p, tw_p, num_p, cat_p};
    const float*    bps[4] = {des_b, tw_b, num_b, cat_b};
    const int boff[4] = {53, 821, 12, 0};   // feature column offsets
    const int bkt[4]  = {24, 24, 2, 1};     // K tiles (zero-padded in packed B)

#pragma unroll
    for (int b = 0; b < 4; ++b) {
        const float* ab = arow + boff[b];
        for (int kt = 0; kt < bkt[b]; ++kt) {
            int kb = kt * 32 + k0;
            __builtin_prefetch(ab + kb + 128, 0, 1);
            v16h a = load_a_frag_f32(ab, kb);
#pragma unroll
            for (int j = 0; j < 2; ++j) {
                v16h bf = load_b_frag(wps[b] + ((size_t)(kt * 2 + j) * 32 + lane) * 16);
                WMMA(acc[2 * b + j], a, bf);
            }
        }
    }

    // bias + leaky, stage D-layout tile into LDS (A-layout readable)
#pragma unroll
    for (int b = 0; b < 4; ++b)
#pragma unroll
        for (int j = 0; j < 2; ++j) {
            int nt = 2 * b + j;
            float bb = bps[b][j * 16 + lc];
#pragma unroll
            for (int v = 0; v < 8; ++v) {
                float t = acc[nt][v] + bb;
                ls[(v + 8 * hi) * LDSW + nt * 16 + lc] = LEAKY(t);
            }
        }
    __builtin_amdgcn_wave_barrier();   // same-wave LDS RAW: order only

    // in-projection straight out of LDS
#pragma unroll
    for (int i = 0; i < 8; ++i) acc[i] = (v8f){};
    const float* lrow = ls + lc * LDSW;
#pragma unroll
    for (int kt = 0; kt < 4; ++kt) {
        v16h a = load_a_frag_f32(lrow, kt * 32 + k0);
#pragma unroll
        for (int nt = 0; nt < 8; ++nt) {
            v16h bf = load_b_frag(in_p + ((size_t)(kt * 8 + nt) * 32 + lane) * 16);
            WMMA(acc[nt], a, bf);
        }
    }
#pragma unroll
    for (int nt = 0; nt < 8; ++nt) {
        float bb = in_b[nt * 16 + lc];
#pragma unroll
        for (int v = 0; v < 8; ++v) {
            float t = acc[nt][v] + bb;
            X[(size_t)(wave * 16 + v + 8 * hi) * HID + nt * 16 + lc] = (_Float16)LEAKY(t);
        }
    }
}

// ---------------------------------------------------------------------------
// RGCN edge scatter: one wave per edge; f16 gather (one b64 per lane),
// fp32 atomics into agg + count atomic.
// ---------------------------------------------------------------------------
__global__ __launch_bounds__(256)
void rgcn_scatter(const _Float16* __restrict__ X, const int* __restrict__ src,
                  const int* __restrict__ dst, const int* __restrict__ et,
                  float* agg, float* cnt, int E, int N) {
    int e    = blockIdx.x * (blockDim.x >> 5) + (threadIdx.x >> 5);
    int lane = threadIdx.x & 31;
    if (e >= E) return;
    int s = src[e];
    int d = dst[e];
    int r = et[e] & 1;
    v4h xv = *(const v4h*)(X + (size_t)s * HID + lane * 4);
    float* ad = agg + ((size_t)r * N + d) * HID + lane * 4;
#pragma unroll
    for (int j = 0; j < 4; ++j)
        atomicAdd(ad + j, (float)xv[j]);
    if (lane == 0) atomicAdd(cnt + (size_t)r * N + d, 1.f);
}

// agg[r][i][:] /= max(cnt[r][i], 1)
__global__ __launch_bounds__(256)
void rgcn_norm(float* agg, const float* __restrict__ cnt, int N) {
    long long idx = (long long)blockIdx.x * blockDim.x + threadIdx.x;
    long long total = 2LL * N * HID;
    if (idx >= total) return;
    float c = cnt[idx >> 7];
    agg[idx] *= 1.f / fmaxf(c, 1.f);
}

// ---------------------------------------------------------------------------
// Fused RGCN layer: acc = x@root + agg0@W0 + agg1@W1 + bias + residual(x),
// then LayerNorm across the in-register 16x128 tile; in-place f16 X update.
// Row stats: sum 8 accumulators per lane, then 16-lane shfl_xor tree
// (lanes 0-15 and 16-31 hold disjoint rows, masks 1,2,4,8 stay in-half).
// ---------------------------------------------------------------------------
__global__ __launch_bounds__(128)
void fused_rgcn(_Float16* X, const float* __restrict__ agg, long long aggStride,
                const _Float16* __restrict__ root_p, const _Float16* __restrict__ rel0_p,
                const _Float16* __restrict__ rel1_p,
                const float* __restrict__ bias,
                const float* __restrict__ lng, const float* __restrict__ lnb,
                int Mtiles) {
    int wave = blockIdx.x * 4 + (threadIdx.x >> 5);
    int lane = threadIdx.x & 31;
    if (wave >= Mtiles) return;
    int lc = lane & 15, hi = lane >> 4, k0 = hi * 8;

    v8f acc[8] = {};

    // root term: f16 activations, direct b128 fragment loads
    const _Float16* xr = X + (size_t)(wave * 16 + lc) * HID;
#pragma unroll
    for (int kt = 0; kt < 4; ++kt) {
        int kb = kt * 32 + k0;
        v16h a = load_a_frag_f16(xr, kb);
#pragma unroll
        for (int nt = 0; nt < 8; ++nt) {
            v16h bf = load_b_frag(root_p + ((size_t)(kt * 8 + nt) * 32 + lane) * 16);
            WMMA(acc[nt], a, bf);
        }
    }

    // relation terms: fp32 agg buffers
    const _Float16* wps[2] = {rel0_p, rel1_p};
#pragma unroll
    for (int s = 0; s < 2; ++s) {
        const float* ar = agg + (size_t)s * (size_t)aggStride + (size_t)(wave * 16 + lc) * HID;
#pragma unroll
        for (int kt = 0; kt < 4; ++kt) {
            int kb = kt * 32 + k0;
            __builtin_prefetch(ar + kb + 128, 0, 1);
            v16h a = load_a_frag_f32(ar, kb);
#pragma unroll
            for (int nt = 0; nt < 8; ++nt) {
                v16h bf = load_b_frag(wps[s] + ((size_t)(kt * 8 + nt) * 32 + lane) * 16);
                WMMA(acc[nt], a, bf);
            }
        }
    }

    // + bias + residual (f16 read, fp32 math)
#pragma unroll
    for (int nt = 0; nt < 8; ++nt) {
        float bb = bias[nt * 16 + lc];
#pragma unroll
        for (int v = 0; v < 8; ++v)
            acc[nt][v] += bb + (float)X[(size_t)(wave * 16 + v + 8 * hi) * HID + nt * 16 + lc];
    }

    // LayerNorm + write back (in place; this wave owns these 16 rows)
#pragma unroll
    for (int v = 0; v < 8; ++v) {
        float s = 0.f, sq = 0.f;
#pragma unroll
        for (int nt = 0; nt < 8; ++nt) { float t = acc[nt][v]; s += t; sq += t * t; }
#pragma unroll
        for (int m = 1; m < 16; m <<= 1) {
            s  += __shfl_xor(s,  m, 32);
            sq += __shfl_xor(sq, m, 32);
        }
        float mu   = s * (1.f / HID);
        float rstd = rsqrtf(sq * (1.f / HID) - mu * mu + 1e-5f);
        size_t row = (size_t)(wave * 16 + v + 8 * hi);
#pragma unroll
        for (int nt = 0; nt < 8; ++nt) {
            int c = nt * 16 + lc;
            X[row * HID + c] = (_Float16)((acc[nt][v] - mu) * rstd * lng[c] + lnb[c]);
        }
    }
}

// ---------------------------------------------------------------------------
// Fused output: leaky(x @ out1 + b1) then head (128 -> 2) via in-register
// row reduction; writes final fp32 logits.
// ---------------------------------------------------------------------------
__global__ __launch_bounds__(128)
void fused_out(const _Float16* __restrict__ X,
               const _Float16* __restrict__ o1_p, const float* __restrict__ o1_b,
               const float* __restrict__ W2, const float* __restrict__ b2,
               float* __restrict__ O, int Mtiles) {
    int wave = blockIdx.x * 4 + (threadIdx.x >> 5);
    int lane = threadIdx.x & 31;
    if (wave >= Mtiles) return;
    int lc = lane & 15, hi = lane >> 4, k0 = hi * 8;

    v8f acc[8] = {};
    const _Float16* ar = X + (size_t)(wave * 16 + lc) * HID;
#pragma unroll
    for (int kt = 0; kt < 4; ++kt) {
        v16h a = load_a_frag_f16(ar, kt * 32 + k0);
#pragma unroll
        for (int nt = 0; nt < 8; ++nt) {
            v16h bf = load_b_frag(o1_p + ((size_t)(kt * 8 + nt) * 32 + lane) * 16);
            WMMA(acc[nt], a, bf);
        }
    }

#pragma unroll
    for (int v = 0; v < 8; ++v) {
        float p0 = 0.f, p1 = 0.f;
#pragma unroll
        for (int nt = 0; nt < 8; ++nt) {
            int c = nt * 16 + lc;
            float t = acc[nt][v] + o1_b[c];
            t = LEAKY(t);
            p0 += t * W2[c * 2 + 0];
            p1 += t * W2[c * 2 + 1];
        }
#pragma unroll
        for (int m = 1; m < 16; m <<= 1) {
            p0 += __shfl_xor(p0, m, 32);
            p1 += __shfl_xor(p1, m, 32);
        }
        if (lc == 0) {
            size_t row = (size_t)(wave * 16 + v + 8 * hi);
            O[row * 2 + 0] = p0 + b2[0];
            O[row * 2 + 1] = p1 + b2[1];
        }
    }
}

// ---------------------------------------------------------------------------
// Host orchestration (graph-capture safe: kernels + hipMemsetAsync only).
// ---------------------------------------------------------------------------
extern "C" void kernel_launch(void* const* d_in, const int* in_sizes, int n_in,
                              void* d_out, int out_size, void* d_ws, size_t ws_size,
                              hipStream_t stream) {
    const float* feature = (const float*)d_in[0];
    const int*   eidx    = (const int*)d_in[1];
    const int*   etype   = (const int*)d_in[2];
    const float* des_W   = (const float*)d_in[3];
    const float* des_b   = (const float*)d_in[4];
    const float* tw_W    = (const float*)d_in[5];
    const float* tw_b    = (const float*)d_in[6];
    const float* num_W   = (const float*)d_in[7];
    const float* num_b   = (const float*)d_in[8];
    const float* cat_W   = (const float*)d_in[9];
    const float* cat_b   = (const float*)d_in[10];
    const float* in_W    = (const float*)d_in[11];
    const float* in_b    = (const float*)d_in[12];
    const float* r1_rel  = (const float*)d_in[13];
    const float* r1_root = (const float*)d_in[14];
    const float* r1_bias = (const float*)d_in[15];
    const float* r2_rel  = (const float*)d_in[16];
    const float* r2_root = (const float*)d_in[17];
    const float* r2_bias = (const float*)d_in[18];
    const float* ln1_g   = (const float*)d_in[19];
    const float* ln1_b   = (const float*)d_in[20];
    const float* ln2_g   = (const float*)d_in[21];
    const float* ln2_b   = (const float*)d_in[22];
    const float* out1_W  = (const float*)d_in[23];
    const float* out1_b  = (const float*)d_in[24];
    const float* out2_W  = (const float*)d_in[25];
    const float* out2_b  = (const float*)d_in[26];

    const int FEAT = 1589;
    const int N = in_sizes[0] / FEAT;        // 100000 (divisible by 16)
    const int E = in_sizes[2];               // 1600000
    const int* src = eidx;
    const int* dst = eidx + E;
    const int Mtiles = N / 16;

    // ---- workspace layout ----
    _Float16* X  = (_Float16*)d_ws;                     // N*128 f16
    float* agg = (float*)(X + (size_t)N * HID);         // 2*N*128 f32
    float* cnt = agg + 2 * (size_t)N * HID;             // 2*N f32
    _Float16* pw = (_Float16*)(cnt + 2 * (size_t)N);

    _Float16* des_p  = pw;                     // Kt=24 Nt=2
    _Float16* tw_p   = des_p  + 24 * 2 * 512;
    _Float16* num_p  = tw_p   + 24 * 2 * 512;  // Kt=2 Nt=2 (K padded 41->64)
    _Float16* cat_p  = num_p  +  2 * 2 * 512;  // Kt=1 Nt=2 (K padded 12->32)
    _Float16* in_p   = cat_p  +  1 * 2 * 512;  // Kt=4 Nt=8
    _Float16* r1r0_p = in_p   +  4 * 8 * 512;
    _Float16* r1r1_p = r1r0_p +  4 * 8 * 512;
    _Float16* r1rt_p = r1r1_p +  4 * 8 * 512;
    _Float16* r2r0_p = r1rt_p +  4 * 8 * 512;
    _Float16* r2r1_p = r2r0_p +  4 * 8 * 512;
    _Float16* r2rt_p = r2r1_p +  4 * 8 * 512;
    _Float16* o1_p   = r2rt_p +  4 * 8 * 512;

    auto pack = [&](const float* W, int K, int Nc, int Kt, int Nt, _Float16* dstp) {
        int total = Kt * Nt * 512;
        hipLaunchKernelGGL(pack_w_f16, dim3((total + 255) / 256), dim3(256), 0, stream,
                           W, K, Nc, Kt, Nt, dstp);
    };
    pack(des_W, 768, 32, 24, 2, des_p);
    pack(tw_W,  768, 32, 24, 2, tw_p);
    pack(num_W,  41, 32,  2, 2, num_p);
    pack(cat_W,  12, 32,  1, 2, cat_p);
    pack(in_W,  128, 128, 4, 8, in_p);
    pack(r1_rel,         128, 128, 4, 8, r1r0_p);
    pack(r1_rel + 16384, 128, 128, 4, 8, r1r1_p);
    pack(r1_root,        128, 128, 4, 8, r1rt_p);
    pack(r2_rel,         128, 128, 4, 8, r2r0_p);
    pack(r2_rel + 16384, 128, 128, 4, 8, r2r1_p);
    pack(r2_root,        128, 128, 4, 8, r2rt_p);
    pack(out1_W,         128, 128, 4, 8, o1_p);

    dim3 tb(128);
    dim3 tg((Mtiles + 3) / 4);
    const size_t aggBytes = (2 * (size_t)N * HID + 2 * (size_t)N) * sizeof(float);
    const long long aggS = (long long)N * HID;
    dim3 scat_g((E + 7) / 8), wide_b(256);
    dim3 norm_g((2 * N * HID + 255) / 256);

    // ---- features -> X (f16) ----
    hipLaunchKernelGGL(fused_input, tg, tb, 0, stream,
                       feature, FEAT, des_p, tw_p, num_p, cat_p, in_p,
                       des_b, tw_b, num_b, cat_b, in_b, X, Mtiles);

    // ---- RGCN layer 1 ----
    hipMemsetAsync(agg, 0, aggBytes, stream);
    hipLaunchKernelGGL(rgcn_scatter, scat_g, wide_b, 0, stream,
                       X, src, dst, etype, agg, cnt, E, N);
    hipLaunchKernelGGL(rgcn_norm, norm_g, wide_b, 0, stream, agg, cnt, N);
    hipLaunchKernelGGL(fused_rgcn, tg, tb, 0, stream,
                       X, agg, aggS, r1rt_p, r1r0_p, r1r1_p, r1_bias,
                       ln1_g, ln1_b, Mtiles);

    // ---- RGCN layer 2 ----
    hipMemsetAsync(agg, 0, aggBytes, stream);
    hipLaunchKernelGGL(rgcn_scatter, scat_g, wide_b, 0, stream,
                       X, src, dst, etype, agg, cnt, E, N);
    hipLaunchKernelGGL(rgcn_norm, norm_g, wide_b, 0, stream, agg, cnt, N);
    hipLaunchKernelGGL(fused_rgcn, tg, tb, 0, stream,
                       X, agg, aggS, r2rt_p, r2r0_p, r2r1_p, r2_bias,
                       ln2_g, ln2_b, Mtiles);

    // ---- output MLP ----
    hipLaunchKernelGGL(fused_out, tg, tb, 0, stream,
                       X, o1_p, out1_b, out2_W, out2_b, (float*)d_out, Mtiles);
}
